// FinalLayer_41704132444473
// MI455X (gfx1250) — matrix operand order repaired and checked by reading, not compiled, not run on hardware
//
#include <hip/hip_runtime.h>

typedef __attribute__((ext_vector_type(16))) _Float16 v16h;
typedef __attribute__((ext_vector_type(8)))  float    v8f;

#define B_    4
#define N_IN  512
#define N_OUT 1024
#define C_    64
#define THREADS 512

#define LOG2E 1.4426950408889634f

// One block = one (batch, 16-m tile). 16 waves = 4 c-tiles x 4 n-quarters.
// Each wave accumulates D[16m x 16c] over 128 n via 64 WMMA steps
// (k = c' + 16*n_i embedding; B is block-diagonal y, A is the RBF weights).
__global__ __launch_bounds__(THREADS) void rbf_wmma_kernel(
    const float* __restrict__ x,      // (B, N_IN)
    const float* __restrict__ y,      // (B, N_IN, C)
    const float* __restrict__ t,      // (B, N_OUT)
    const float* __restrict__ sigma,  // (C)
    const float* __restrict__ w,      // (C, 1)
    const float* __restrict__ bias,   // (1)
    float* __restrict__ out)          // (B, N_OUT, 1)
{
    __shared__ unsigned int syu[N_IN * (C_ / 2)];  // y as packed f16 pairs, [n][c/2] : 64 KB
    __shared__ float sout[16];

    const int tid  = threadIdx.x;
    const int lane = tid & 31;
    const int wave = tid >> 5;

    const int bidx = blockIdx.x;
    const int b    = bidx >> 6;            // N_OUT/16 = 64 m-tiles per batch
    const int m0   = (bidx & 63) << 4;

    const float* xb = x + b * N_IN;
    const float* yb = y + (size_t)b * N_IN * C_;

    // hint: pull the y panel toward the caches early
    __builtin_prefetch(yb + tid * 64, 0, 0);

    // ---- stage y: f32 -> packed f16 pairs in LDS ----
    {
        const float2* y2 = (const float2*)yb;
        for (int i = tid; i < N_IN * (C_ / 2); i += THREADS) {
            float2 f = y2[i];
            syu[i] = __builtin_bit_cast(unsigned int,
                         __builtin_amdgcn_cvt_pkrtz(f.x, f.y));
        }
    }
    if (tid < 16) sout[tid] = 0.0f;
    __syncthreads();

    // ---- per-wave / per-lane setup ----
    const int ct    = wave & 3;        // c-tile of 16 channels
    const int nq    = wave >> 2;       // n-quarter: 128 n each
    const int cbase = ct << 4;
    const int chalf = lane >> 4;       // 0: lanes 0-15, 1: lanes 16-31
    const int mrow  = lane & 15;       // A row (m) == B column (c) index

    const float tm = t[b * N_OUT + m0 + mrow];

    // A-side channels for this lane: c' = j, j = 0..15  (matches K layout)
    // wt = exp(-0.5*d/s^2) = exp2(d * q),  q = -0.5*log2e*exp(-2*sigma)
    float q[16];
    #pragma unroll
    for (int j = 0; j < 16; ++j) {
        float sg = sigma[cbase + j];
        q[j] = -0.5f * LOG2E * __builtin_amdgcn_exp2f(-2.0f * sg * LOG2E);
    }

    // B-side: lane holds column c = cbase+mrow; single nonzero at half index
    // p = mrow inside its 16-half K group. Note (cB & 1) == (lane & 1), so the
    // loaded y-pair word already has the value in the correct 16-bit half.
    const int cB  = cbase + mrow;
    const int sh  = (lane & 1) * 16;
    const unsigned int hmask = 0xFFFFu << sh;
    const int p2 = mrow >> 1;
    unsigned int bmask[8];
    #pragma unroll
    for (int j = 0; j < 8; ++j) bmask[j] = (j == p2) ? hmask : 0u;

    const int   ywordcol = cB >> 1;
    const float pw       = w[cB];

    v8f acc = {};
    const int nbase0 = nq * (N_IN / 4);

    #pragma unroll 2
    for (int step = 0; step < N_IN / 8; ++step) {
        const int n0 = nbase0 + step * 2;
        const int n1 = n0 + 1;

        // squared distances (x uniform per wave -> scalar loads; t per lane)
        float d0 = xb[n0] - tm; d0 *= d0;
        float d1 = xb[n1] - tm; d1 *= d1;

        // ---- A: RBF weights, 16 f16 per lane (lanes 0-15: n0, lanes 16-31: n1; h = channel) ----
        union { v16h v; unsigned int u[8]; } A;
        const float dh = chalf ? d1 : d0;
        #pragma unroll
        for (int j = 0; j < 8; ++j) {
            float e0 = __builtin_amdgcn_exp2f(dh * q[2 * j]);
            float e1 = __builtin_amdgcn_exp2f(dh * q[2 * j + 1]);
            A.u[j] = __builtin_bit_cast(unsigned int,
                         __builtin_amdgcn_cvt_pkrtz(e0, e1));
        }

        // ---- B: block-diagonal embedding of y (1 LDS load + 8 ANDs) ----
        const int nl = n0 + chalf;               // lanes 0-15 -> n0, 16-31 -> n1
        const unsigned int yword = syu[nl * (C_ / 2) + ywordcol];
        union { v16h v; unsigned int u[8]; } Bm;
        #pragma unroll
        for (int j = 0; j < 8; ++j) Bm.u[j] = yword & bmask[j];

        acc = __builtin_amdgcn_wmma_f32_16x16x32_f16(
            false, A.v, false, Bm.v, (short)0, acc, false, false);
    }

    // ---- projection + combine: out[m] = bias + sum_c D[m,c]*w[c] ----
    // D layout: lane L, VGPR r -> (m = r + 8*(L>>4), c = L&15).
    // Reduce over c with a 4-step shuffle butterfly inside each 16-lane half
    // (xor masks 1,2,4,8 never cross the half boundary), then one LDS atomic
    // per (wave, m) to combine the 16 waves.
    #pragma unroll
    for (int r = 0; r < 8; ++r) {
        float v = acc[r] * pw;
        v += __shfl_xor(v, 1);
        v += __shfl_xor(v, 2);
        v += __shfl_xor(v, 4);
        v += __shfl_xor(v, 8);
        if (mrow == 0) atomicAdd(&sout[r + 8 * chalf], v);
    }
    __syncthreads();

    if (tid < 16) {
        out[b * N_OUT + m0 + tid] = sout[tid] + bias[0];
    }
}

extern "C" void kernel_launch(void* const* d_in, const int* in_sizes, int n_in,
                              void* d_out, int out_size, void* d_ws, size_t ws_size,
                              hipStream_t stream) {
    const float* x     = (const float*)d_in[0];
    const float* y     = (const float*)d_in[1];
    const float* t     = (const float*)d_in[2];
    const float* sigma = (const float*)d_in[3];
    const float* w     = (const float*)d_in[4];
    const float* bias  = (const float*)d_in[5];
    float* out = (float*)d_out;

    dim3 grid(B_ * (N_OUT / 16));   // 256 blocks: (batch, m-tile)
    dim3 block(THREADS);            // 16 waves (wave32)
    hipLaunchKernelGGL(rbf_wmma_kernel, grid, block, 0, stream,
                       x, y, t, sigma, w, bias, out);
}
